// HierarchicalFFT_37469294690507
// MI455X (gfx1250) — matrix operand order repaired and checked by reading, not compiled, not run on hardware
//
#include <hip/hip_runtime.h>
#include <math.h>

#define B_      2
#define S_      65536
#define D_      256
#define C_      10
#define CHUNK_  8192
#define HOP_    7168
#define N2_     4096
#define FB_     4097

typedef __attribute__((ext_vector_type(16))) _Float16 v16h;
typedef __attribute__((ext_vector_type(8)))  float    v8f;

// ---------------------------------------------------------------------------
// K1: window + 8192-pt real FFT (as 4096-pt complex radix-2 DIT in LDS) per
// (b, chunk, d). Writes cf (interleaved complex) into d_out and feats means
// into workspace. One workgroup (256 threads = 8 wave32) per FFT.
// Twiddles come from a 2048-entry LDS table (one sincos sweep per block).
// ---------------------------------------------------------------------------
__global__ __launch_bounds__(256)
void hfft_window_fft_kernel(const float* __restrict__ x,
                            const float* __restrict__ ww,
                            const float* __restrict__ fsc,
                            float* __restrict__ out,     // complex interleaved
                            float* __restrict__ feats)   // [re 5120][im 5120]
{
    __shared__ float2 z[N2_];         // 32 KB
    __shared__ float2 tw[N2_ / 2];    // 16 KB: tw[j] = e^{-2*pi*i*j/4096}
    __shared__ float2 red[256];       // 2 KB

    const int tid = threadIdx.x;
    const int blk = blockIdx.x;            // b*C*D + c*D + d
    const int d   = blk % D_;
    const int c   = (blk / D_) % C_;
    const int b   = blk / (D_ * C_);

    // softmax of 3 window weights (scalar, uniform)
    float a0 = ww[0], a1 = ww[1], a2 = ww[2];
    float mx = fmaxf(a0, fmaxf(a1, a2));
    float e0 = __expf(a0 - mx), e1 = __expf(a1 - mx), e2 = __expf(a2 - mx);
    float inv = 1.0f / (e0 + e1 + e2);
    float sw0 = e0 * inv, sw1 = e1 * inv, sw2 = e2 * inv;
    // window(s) = wA - wB*cos(2*pi*s/8192)
    const float wA = 0.5f * sw0 + 0.54f * sw1 + sw2;
    const float wB = 0.5f * sw0 + 0.46f * sw1;
    const float W0 = 7.669903939428206e-4f;    // 2*pi/8192
    const float T0 = -1.5339807878856412e-3f;  // -2*pi/4096

    // twiddle table
    for (int j = tid; j < N2_ / 2; j += 256) {
        float sn, cs;
        __sincosf(T0 * (float)j, &sn, &cs);
        tw[j] = make_float2(cs, sn);
    }

    // load pairs z[n] = win(2n)*x[2n] + i*win(2n+1)*x[2n+1], bit-reversed
    const long baseS = (long)b * S_ * D_ + (long)c * HOP_ * D_ + d;
    const int  smax  = S_ - c * HOP_;          // valid samples in this chunk
    for (int n = tid; n < N2_; n += 256) {
        int s0 = 2 * n, s1 = 2 * n + 1;
        float x0 = (s0 < smax) ? x[baseS + (long)s0 * D_] : 0.0f;
        float x1 = (s1 < smax) ? x[baseS + (long)s1 * D_] : 0.0f;
        float c0 = __cosf(W0 * (float)s0);
        float c1 = __cosf(W0 * (float)s1);
        int rev = (int)(__brev((unsigned)n) >> 20);   // 12-bit reversal
        z[rev] = make_float2(x0 * (wA - wB * c0), x1 * (wA - wB * c1));
    }
    __syncthreads();

    // 12 radix-2 DIT stages in LDS
    for (int m = 1; m <= 12; ++m) {
        const int hlf = 1 << (m - 1);
        const int sh  = 12 - m;
        for (int j = tid; j < (N2_ >> 1); j += 256) {
            int idx  = j & (hlf - 1);
            int base = ((j >> (m - 1)) << m) + idx;
            float2 w = tw[idx << sh];
            float2 u = z[base];
            float2 t = z[base + hlf];
            float2 v = make_float2(t.x * w.x - t.y * w.y, t.x * w.y + t.y * w.x);
            z[base]       = make_float2(u.x + v.x, u.y + v.y);
            z[base + hlf] = make_float2(u.x - v.x, u.y - v.y);
        }
        __syncthreads();
    }

    // Hermitian unpack to 4097 rfft bins, ortho norm, freq_scales, store + sum
    const float invsq = 0.011048543456039805f;   // 1/sqrt(8192)
    float2* outc = (float2*)out;
    const long obase = (long)(b * C_ + c) * FB_ * D_ + d;
    float sr = 0.0f, si = 0.0f;
    for (int k = tid; k <= N2_; k += 256) {
        float2 za = z[k & (N2_ - 1)];
        float2 zb = z[(N2_ - k) & (N2_ - 1)];
        float er = 0.5f * (za.x + zb.x);
        float ei = 0.5f * (za.y - zb.y);
        float dr = za.x - zb.x;
        float di = za.y + zb.y;
        float orr = 0.5f * di;      // Xo = (a - conj(b)) / (2i)
        float oii = -0.5f * dr;
        float sn, cs;
        __sincosf(W0 * (float)k, &sn, &cs);     // tw8192 = cs - i*sn
        float xr = er + cs * orr + sn * oii;
        float xi = ei + cs * oii - sn * orr;
        float sc = fsc[k] * invsq;
        xr *= sc; xi *= sc;
        sr += xr; si += xi;
        outc[obase + (long)k * D_] = make_float2(xr, xi);
    }
    __syncthreads();

    // block reduction for feats mean
    red[tid] = make_float2(sr, si);
    __syncthreads();
    for (int off = 128; off > 0; off >>= 1) {
        if (tid < off) {
            red[tid].x += red[tid + off].x;
            red[tid].y += red[tid + off].y;
        }
        __syncthreads();
    }
    if (tid == 0) {
        const float invF = 1.0f / (float)FB_;
        feats[(b * C_ + c) * D_ + d]        = red[0].x * invF;
        feats[5120 + (b * C_ + c) * D_ + d] = red[0].y * invF;
    }
}

// ---------------------------------------------------------------------------
// Shared GEMM tile: Out[M x 256] = A[M x 256] @ W^T, one wave32 per 16x16
// output tile, fully unrolled K with v_wmma_f32_16x16x32_f16.
// Rows >= Mrows are clamped to row 0 on load (they only feed accumulator rows
// that are never stored), so the hot loop is branch-free b128 loads + cvt.
// ---------------------------------------------------------------------------
__device__ __forceinline__ void gemm_awt_tile(const float* __restrict__ A,
                                              const float* __restrict__ W,
                                              float* __restrict__ Out,
                                              int Mrows, int lane,
                                              int mt, int nt)
{
    const int lh  = (lane < 16) ? 0 : 1;
    const int l16 = lane & 15;
    const int M   = mt * 16 + l16;
    const int Mc  = (M < Mrows) ? M : 0;   // clamp; extra rows never stored
    const float* arow = A + Mc * 256;
    const float* brow = W + (nt * 16 + l16) * 256;

    v8f acc = {};
#pragma unroll
    for (int k0 = 0; k0 < 256; k0 += 32) {
        const float4* pa0 = (const float4*)(arow + k0 + lh * 8);        // K: +0..7
        const float4* pa1 = (const float4*)(arow + k0 + 16 + lh * 8);   // K: +16..23
        const float4* pb  = (const float4*)(brow + k0 + lh * 8);        // K: +0..7
        const float4* pb1 = (const float4*)(brow + k0 + 16 + lh * 8);   // K: +16..23
        float4 a0 = pa0[0], a1 = pa0[1];
        float4 a2 = pa1[0], a3 = pa1[1];
        float4 b0 = pb[0], b1 = pb[1], b2 = pb1[0], b3 = pb1[1];
        v16h af, bf;
        af[0]  = (_Float16)a0.x; af[1]  = (_Float16)a0.y;
        af[2]  = (_Float16)a0.z; af[3]  = (_Float16)a0.w;
        af[4]  = (_Float16)a1.x; af[5]  = (_Float16)a1.y;
        af[6]  = (_Float16)a1.z; af[7]  = (_Float16)a1.w;
        af[8]  = (_Float16)a2.x; af[9]  = (_Float16)a2.y;
        af[10] = (_Float16)a2.z; af[11] = (_Float16)a2.w;
        af[12] = (_Float16)a3.x; af[13] = (_Float16)a3.y;
        af[14] = (_Float16)a3.z; af[15] = (_Float16)a3.w;
        bf[0]  = (_Float16)b0.x; bf[1]  = (_Float16)b0.y;
        bf[2]  = (_Float16)b0.z; bf[3]  = (_Float16)b0.w;
        bf[4]  = (_Float16)b1.x; bf[5]  = (_Float16)b1.y;
        bf[6]  = (_Float16)b1.z; bf[7]  = (_Float16)b1.w;
        bf[8]  = (_Float16)b2.x; bf[9]  = (_Float16)b2.y;
        bf[10] = (_Float16)b2.z; bf[11] = (_Float16)b2.w;
        bf[12] = (_Float16)b3.x; bf[13] = (_Float16)b3.y;
        bf[14] = (_Float16)b3.z; bf[15] = (_Float16)b3.w;
        acc = __builtin_amdgcn_wmma_f32_16x16x32_f16(
                  false, af, false, bf, (short)0, acc, false, false);
    }
#pragma unroll
    for (int j = 0; j < 8; ++j) {
        int Mo = mt * 16 + j + lh * 8;
        if (Mo < Mrows) Out[Mo * 256 + nt * 16 + l16] = acc[j];
    }
}

// K2a: all six Q/K/V projections in one launch.
// grid = (32, 6): x -> (mtile, ntile); y -> (component re/im, weight q/k/v)
__global__ __launch_bounds__(32)
void hfft_gemm_qkv_wmma(const float* __restrict__ feats,
                        const float* __restrict__ Wq,
                        const float* __restrict__ Wk,
                        const float* __restrict__ Wv,
                        float* __restrict__ Qo,
                        float* __restrict__ Ko,
                        float* __restrict__ Vo)
{
    const int lane = threadIdx.x;
    const int nt = blockIdx.x & 15;
    const int mt = (blockIdx.x >> 4) & 1;
    const int comp  = blockIdx.y & 1;
    const int which = blockIdx.y >> 1;
    const float* A = feats + comp * 5120;
    const float* W = (which == 0) ? Wq : (which == 1) ? Wk : Wv;
    float* Out = (((which == 0) ? Qo : (which == 1) ? Ko : Vo)) + comp * 5120;
    gemm_awt_tile(A, W, Out, 20, lane, mt, nt);
}

// K2b: output projection (re + im) in one launch. grid = (32, 2)
__global__ __launch_bounds__(32)
void hfft_gemm_proj_wmma(const float* __restrict__ ctx,
                         const float* __restrict__ Wo,
                         float* __restrict__ proj)
{
    const int lane = threadIdx.x;
    const int nt = blockIdx.x & 15;
    const int mt = (blockIdx.x >> 4) & 1;
    const int comp = blockIdx.y;
    gemm_awt_tile(ctx + comp * 5120, Wo, proj + comp * 5120, 20, lane, mt, nt);
}

// ---------------------------------------------------------------------------
// K3: complex scores -> complex sigmoid gates -> ctx = gates @ V.
// Tiny (2x10x10 scores, 20x256 ctx); single block.
// Q/K/V/ctx layout: [re 5120][im 5120] floats.
// ---------------------------------------------------------------------------
__global__ __launch_bounds__(256)
void hfft_attn_small_kernel(const float* __restrict__ Q,
                            const float* __restrict__ Km,
                            const float* __restrict__ V,
                            float* __restrict__ ctx)
{
    __shared__ float g_re[200];
    __shared__ float g_im[200];
    const int t = threadIdx.x;

    if (t < 200) {
        int b  = t / 100;
        int ce = t % 100;
        int cc = ce / 10, ee = ce % 10;
        const float* qr = Q + (b * 10 + cc) * 256;
        const float* qi = Q + 5120 + (b * 10 + cc) * 256;
        const float* kr = Km + (b * 10 + ee) * 256;
        const float* ki = Km + 5120 + (b * 10 + ee) * 256;
        float sr = 0.0f, si = 0.0f;
        for (int dd = 0; dd < 256; ++dd) {
            sr += qr[dd] * kr[dd] - qi[dd] * ki[dd];
            si += qr[dd] * ki[dd] + qi[dd] * kr[dd];
        }
        sr *= 0.0625f;  si *= 0.0625f;      // / sqrt(256)
        // complex sigmoid: 1/(1+exp(-z)), exp(-z) = e^{-sr}(cos si - i sin si)
        float ex = __expf(-sr);
        float wr = ex * __cosf(si);
        float wi = -ex * __sinf(si);
        float dr  = 1.0f + wr;
        float den = dr * dr + wi * wi;
        g_re[t] = dr  / den;
        g_im[t] = -wi / den;
    }
    __syncthreads();

    for (int i = t; i < 2 * 10 * 256; i += 256) {
        int b  = i / 2560;
        int cd = i % 2560;
        int cc = cd / 256, dd = cd % 256;
        float cr = 0.0f, ci = 0.0f;
#pragma unroll
        for (int ee = 0; ee < 10; ++ee) {
            float gr = g_re[b * 100 + cc * 10 + ee];
            float gi = g_im[b * 100 + cc * 10 + ee];
            float vr = V[(b * 10 + ee) * 256 + dd];
            float vi = V[5120 + (b * 10 + ee) * 256 + dd];
            cr += gr * vr - gi * vi;
            ci += gr * vi + gi * vr;
        }
        ctx[(b * 10 + cc) * 256 + dd]        = cr;
        ctx[5120 + (b * 10 + cc) * 256 + dd] = ci;
    }
}

// ---------------------------------------------------------------------------
// K4: fused = cf + 0.1 * context (broadcast over freq bins), in-place on out.
// ---------------------------------------------------------------------------
__global__ __launch_bounds__(256)
void hfft_fuse_kernel(float* __restrict__ out, const float* __restrict__ proj)
{
    const long total = (long)B_ * C_ * FB_ * D_;
    long i = (long)blockIdx.x * 256 + threadIdx.x;
    if (i >= total) return;
    int dd = (int)(i % D_);
    int bc = (int)((i / D_) / FB_);
    float2* o = (float2*)out;
    float2 v = o[i];
    v.x += 0.1f * proj[bc * 256 + dd];
    v.y += 0.1f * proj[5120 + bc * 256 + dd];
    o[i] = v;
}

// ---------------------------------------------------------------------------
extern "C" void kernel_launch(void* const* d_in, const int* in_sizes, int n_in,
                              void* d_out, int out_size, void* d_ws, size_t ws_size,
                              hipStream_t stream) {
    (void)in_sizes; (void)n_in; (void)out_size; (void)ws_size;
    const float* x   = (const float*)d_in[0];
    const float* Wq  = (const float*)d_in[1];
    const float* Wk  = (const float*)d_in[2];
    const float* Wv  = (const float*)d_in[3];
    const float* Wo  = (const float*)d_in[4];
    const float* ww  = (const float*)d_in[5];
    const float* fsc = (const float*)d_in[6];
    float* out = (float*)d_out;
    float* ws  = (float*)d_ws;

    // workspace layout (floats): each complex matrix is [re 5120][im 5120]
    float* feats = ws;               // 10240
    float* Q     = ws + 10240;       // 10240
    float* K     = ws + 20480;       // 10240
    float* V     = ws + 30720;       // 10240
    float* ctx   = ws + 40960;       // 10240
    float* proj  = ws + 51200;       // 10240

    // 1) window + FFT + feats (writes all of d_out as pre-fusion cf)
    hfft_window_fft_kernel<<<B_ * C_ * D_, 256, 0, stream>>>(x, ww, fsc, out, feats);

    // 2) Q/K/V projections (re and im separately; weights are real) via WMMA
    hfft_gemm_qkv_wmma<<<dim3(32, 6), 32, 0, stream>>>(feats, Wq, Wk, Wv, Q, K, V);

    // 3) complex gated attention (tiny)
    hfft_attn_small_kernel<<<1, 256, 0, stream>>>(Q, K, V, ctx);

    // 4) output projection via WMMA
    hfft_gemm_proj_wmma<<<dim3(32, 2), 32, 0, stream>>>(ctx, Wo, proj);

    // 5) fuse broadcast context into cf
    const long total = (long)B_ * C_ * FB_ * D_;   // 20,976,640 complex elems
    const int blocks = (int)((total + 255) / 256); // 81,940
    hfft_fuse_kernel<<<blocks, 256, 0, stream>>>(out, proj);
}
